// GPT_26551487823905
// MI455X (gfx1250) — matrix operand order repaired and checked by reading, not compiled, not run on hardware
//
#include <hip/hip_runtime.h>
#include <hip/hip_bf16.h>

// ---------------- model constants ----------------
constexpr int L_  = 4;
constexpr int H_  = 8;
constexpr int D_  = 512;
constexpr int DH_ = 64;
constexpr int T_  = 2048;
constexpr int B_  = 2;
constexpr int V_  = 32000;
constexpr int FF_ = 2048;
constexpr int BT_ = B_ * T_;              // 4096 rows of activations

typedef __attribute__((ext_vector_type(16))) __bf16 v16bf;
typedef __attribute__((ext_vector_type(8)))  float  v8f;

struct __attribute__((aligned(16))) U4 { unsigned x, y, z, w; };

#define WMMA_BF16(a, b, c) \
  __builtin_amdgcn_wmma_f32_16x16x32_bf16(false, (a), false, (b), (short)0, (c), false, false)

__device__ __forceinline__ unsigned short f2bf_u(float f) {
  unsigned u = __builtin_bit_cast(unsigned, f);
  u += 0x7fffu + ((u >> 16) & 1u);      // round-to-nearest-even
  return (unsigned short)(u >> 16);
}

// reduce across the 16 lanes of a wave32 half (lanes 0-15 / 16-31)
__device__ __forceinline__ float half_max16(float v) {
  for (int off = 8; off; off >>= 1) v = fmaxf(v, __shfl_xor(v, off, 32));
  return v;
}
__device__ __forceinline__ float half_sum16(float v) {
  for (int off = 8; off; off >>= 1) v += __shfl_xor(v, off, 32);
  return v;
}

// ---------------- fragment loaders (gfx1250 16-bit WMMA VGPR layouts) ------
// A (16xK, row-major, lda elems): lane m=lane&15 is row; half selects K group.
// VGPR i holds K pair: (i<4 ? 2i : 2i+8) + 8*half  -> two contiguous 16B runs.
__device__ __forceinline__ void load_a_frag(unsigned* u, const unsigned short* Arow,
                                            int k0, int half) {
  U4 lo = *(const U4*)(Arow + k0 + 8 * half);
  U4 hi = *(const U4*)(Arow + k0 + 16 + 8 * half);
  u[0] = lo.x; u[1] = lo.y; u[2] = lo.z; u[3] = lo.w;
  u[4] = hi.x; u[5] = hi.y; u[6] = hi.z; u[7] = hi.w;
}
// B (KxN) supplied TRANSPOSED as [N,K] row-major (ldbt elems): lane n=lane&15
// is col; VGPR i holds K pair k0+(i<4 ? 2i : 2i+8)+8*half -> two contiguous 16B runs.
__device__ __forceinline__ void load_bt_frag(unsigned* u, const unsigned short* Brow,
                                             int k0, int half) {
  U4 lo = *(const U4*)(Brow + k0 + 8 * half);
  U4 hi = *(const U4*)(Brow + k0 + 16 + 8 * half);
  u[0] = lo.x; u[1] = lo.y; u[2] = lo.z; u[3] = lo.w;
  u[4] = hi.x; u[5] = hi.y; u[6] = hi.z; u[7] = hi.w;
}

// ---------------- embedding: x = tok_emb[ctx] + pos_emb --------------------
__global__ void embed_kernel(const int* __restrict__ ctx,
                             const float* __restrict__ tok,
                             const float* __restrict__ pos,
                             float* __restrict__ x) {
  long i = (long)blockIdx.x * blockDim.x + threadIdx.x;
  if (i >= (long)BT_ * D_) return;
  int d = (int)(i % D_);
  long bt = i / D_;
  int t = (int)(bt % T_);
  int b = (int)(bt / T_);
  int id = ctx[b * T_ + t];
  x[i] = tok[(long)id * D_ + d] + pos[(long)t * D_ + d];
}

// ---------------- batched convert f32 -> bf16 with transpose ---------------
// src: [batch, R, C] f32 row-major ; dst: [batch, C, R] bf16 row-major
__global__ void convT_kernel(const float* __restrict__ src,
                             unsigned short* __restrict__ dst,
                             int R, int C, long total) {
  long i = (long)blockIdx.x * blockDim.x + threadIdx.x;
  if (i >= total) return;
  long rc = (long)R * C;
  long bidx = i / rc;
  long rem  = i - bidx * rc;
  int r = (int)(rem / C);
  int c = (int)(rem - (long)r * C);
  dst[bidx * rc + (long)c * R + r] = f2bf_u(src[i]);
}

// ---------------- LayerNorm (one block per row, D=512) ---------------------
__global__ void ln_kernel(const float* __restrict__ x,
                          const float* __restrict__ g,
                          const float* __restrict__ bta,
                          unsigned short* __restrict__ out) {
  __shared__ float red[256];
  const int row = blockIdx.x, tid = threadIdx.x;
  const float* xr = x + (long)row * D_;
  float v0 = xr[tid], v1 = xr[tid + 256];
  red[tid] = v0 + v1;
  __syncthreads();
  for (int s = 128; s; s >>= 1) { if (tid < s) red[tid] += red[tid + s]; __syncthreads(); }
  float mean = red[0] * (1.0f / D_);
  __syncthreads();
  float d0 = v0 - mean, d1 = v1 - mean;
  red[tid] = d0 * d0 + d1 * d1;
  __syncthreads();
  for (int s = 128; s; s >>= 1) { if (tid < s) red[tid] += red[tid + s]; __syncthreads(); }
  float rstd = rsqrtf(red[0] * (1.0f / D_) + 1e-5f);
  unsigned short* o = out + (long)row * D_;
  o[tid]       = f2bf_u(d0 * rstd * g[tid] + bta[tid]);
  o[tid + 256] = f2bf_u(d1 * rstd * g[tid + 256] + bta[tid + 256]);
}

// ---------------- generic WMMA bf16 GEMM -----------------------------------
// C[m,n] = sum_k A[m,k]*B[k,n] (+bias)(+relu)(+accumulate-into-outF)
// A: [M,K] bf16 row-major. Bt: [N,K] bf16 row-major (pre-transposed B).
// Output address = b*out_b_stride + t*out_t_stride + head*out_h_stride + n,
// with m decomposed as b = m / T_rows, t = m % T_rows (head = blockIdx.z).
__global__ void gemm_bf16_wmma(const unsigned short* __restrict__ A, int lda,
                               const unsigned short* __restrict__ Bt, int ldbt,
                               long b_head_stride,
                               const float* __restrict__ bias, int bias_head_stride,
                               float* __restrict__ outF,
                               unsigned short* __restrict__ outB,
                               long out_b_stride, long out_t_stride, long out_h_stride,
                               int M, int N, int K, int T_rows, int relu, int accum) {
  const int lane = threadIdx.x & 31;
  const int tile_n = blockIdx.x * blockDim.y + threadIdx.y;
  const int tile_m = blockIdx.y;
  const int head = blockIdx.z;
  if (tile_n * 16 >= N) return;
  const int col0 = tile_n * 16, row0 = tile_m * 16;
  const int m16 = lane & 15, hf = lane >> 4;

  const unsigned short* Arow = A + (long)(row0 + m16) * lda;
  const unsigned short* Brow = Bt + (long)head * b_head_stride + (long)(col0 + m16) * ldbt;

  v8f c = {};
  for (int k0 = 0; k0 < K; k0 += 32) {
    union { v16bf v; unsigned u[8]; } a, b;
    load_a_frag(a.u, Arow, k0, hf);
    load_bt_frag(b.u, Brow, k0, hf);
    c = WMMA_BF16(a.v, b.v, c);
  }

#pragma unroll
  for (int i = 0; i < 8; ++i) {
    int rm = row0 + i + 8 * hf;
    if (rm >= M) continue;
    int bb = rm / T_rows, tt = rm - bb * T_rows;
    long off = (long)bb * out_b_stride + (long)tt * out_t_stride +
               (long)head * out_h_stride + col0 + m16;
    float acc = c[i];
    if (bias) acc += bias[head * bias_head_stride + col0 + m16];
    if (relu) acc = fmaxf(acc, 0.0f);
    if (outF) { if (accum) outF[off] += acc; else outF[off] = acc; }
    if (outB) outB[off] = f2bf_u(acc);
  }
}

// ---------------- flash attention (one wave per 16 Q-rows per (b,h)) -------
// q,k,v: bf16 [B,H,T,DH] ; accumulates softmax(QK^T/sqrt(dh))V into x [B,T,D].
__global__ void attn_kernel(const unsigned short* __restrict__ Q,
                            const unsigned short* __restrict__ Kw,
                            const unsigned short* __restrict__ Vw,
                            float* __restrict__ X) {
  __shared__ unsigned short pls[16 * 32];          // P tile staging (C->A layout)
  const float SCALE = 0.125f;                      // 1/sqrt(64)
  const int lane = threadIdx.x & 31;
  const int m16 = lane & 15, hf = lane >> 4;
  const int row0 = blockIdx.x * 16;
  const int bh = blockIdx.y;
  const int b = bh / H_, h = bh % H_;
  const unsigned short* q = Q + (long)bh * T_ * DH_;
  const unsigned short* k = Kw + (long)bh * T_ * DH_;
  const unsigned short* v = Vw + (long)bh * T_ * DH_;

  union Frag { v16bf v; unsigned u[8]; };
  Frag aq0, aq1;
  const unsigned short* qrow = q + (long)(row0 + m16) * DH_;
  load_a_frag(aq0.u, qrow, 0, hf);
  load_a_frag(aq1.u, qrow, 32, hf);

  v8f o0 = {}, o1 = {}, o2 = {}, o3 = {};
  float mr[8], lr[8];
#pragma unroll
  for (int i = 0; i < 8; ++i) { mr[i] = -1e30f; lr[i] = 0.0f; }

  for (int c = 0; c <= row0 + 15; c += 32) {       // causal: chunks of 32 keys
    v8f s0 = {}, s1 = {};
    {
      Frag kb;                                     // K^T fragments load contiguously
      const unsigned short* k0r = k + (long)(c + m16) * DH_;
      const unsigned short* k1r = k + (long)(c + 16 + m16) * DH_;
      load_bt_frag(kb.u, k0r, 0, hf);  s0 = WMMA_BF16(aq0.v, kb.v, s0);
      load_bt_frag(kb.u, k0r, 32, hf); s0 = WMMA_BF16(aq1.v, kb.v, s0);
      load_bt_frag(kb.u, k1r, 0, hf);  s1 = WMMA_BF16(aq0.v, kb.v, s1);
      load_bt_frag(kb.u, k1r, 32, hf); s1 = WMMA_BF16(aq1.v, kb.v, s1);
    }
#pragma unroll
    for (int i = 0; i < 8; ++i) {                  // online softmax per row
      int rg = row0 + i + 8 * hf;
      float x0 = s0[i] * SCALE; if (c + m16 > rg)      x0 = -1e30f;
      float x1 = s1[i] * SCALE; if (c + 16 + m16 > rg) x1 = -1e30f;
      float tm = half_max16(fmaxf(x0, x1));
      float mnew = fmaxf(mr[i], tm);
      float fac = __expf(mr[i] - mnew);
      mr[i] = mnew;
      float p0 = __expf(x0 - mnew);
      float p1 = __expf(x1 - mnew);
      lr[i] = lr[i] * fac + half_sum16(p0 + p1);
      o0[i] *= fac; o1[i] *= fac; o2[i] *= fac; o3[i] *= fac;
      int prow = i + 8 * hf;
      pls[prow * 32 + m16]      = f2bf_u(p0);
      pls[prow * 32 + 16 + m16] = f2bf_u(p1);
    }
    __syncthreads();
    Frag pf;                                       // re-swizzle P: C-layout -> A-layout
    load_a_frag(pf.u, &pls[m16 * 32], 0, hf);
    {
      Frag vb;                                     // V is row-major KxN: strided pairs
#pragma unroll
      for (int t = 0; t < 4; ++t) {
        const unsigned short* vbase = v + (long)c * DH_ + t * 16;
#pragma unroll
        for (int i = 0; i < 8; ++i) {
          int kr = (i < 4 ? 2 * i : 2 * i + 8) + 8 * hf;
          unsigned lo = vbase[(long)kr * DH_ + m16];
          unsigned hi = vbase[(long)(kr + 1) * DH_ + m16];
          vb.u[i] = lo | (hi << 16);
        }
        if (t == 0) o0 = WMMA_BF16(pf.v, vb.v, o0);
        if (t == 1) o1 = WMMA_BF16(pf.v, vb.v, o1);
        if (t == 2) o2 = WMMA_BF16(pf.v, vb.v, o2);
        if (t == 3) o3 = WMMA_BF16(pf.v, vb.v, o3);
      }
    }
    __syncthreads();
  }

#pragma unroll
  for (int i = 0; i < 8; ++i) {                    // x += O / l  (residual add)
    int rg = row0 + i + 8 * hf;
    float inv = 1.0f / lr[i];
    long ro = ((long)b * T_ + rg) * D_ + h * DH_ + m16;
    X[ro + 0]  += o0[i] * inv;
    X[ro + 16] += o1[i] * inv;
    X[ro + 32] += o2[i] * inv;
    X[ro + 48] += o3[i] * inv;
  }
}

// ---------------- host side ------------------------------------------------
static inline size_t alignup(size_t o) { return (o + 255) & ~(size_t)255; }

extern "C" void kernel_launch(void* const* d_in, const int* in_sizes, int n_in,
                              void* d_out, int out_size, void* d_ws, size_t ws_size,
                              hipStream_t stream) {
  (void)in_sizes; (void)n_in; (void)out_size; (void)ws_size;
  const int*   ctx    = (const int*)d_in[0];
  const float* tok    = (const float*)d_in[1];
  const float* pos    = (const float*)d_in[2];
  const float* ln1_s  = (const float*)d_in[3];
  const float* ln1_b  = (const float*)d_in[4];
  const float* wq     = (const float*)d_in[5];
  const float* bq     = (const float*)d_in[6];
  const float* wk     = (const float*)d_in[7];
  const float* bk     = (const float*)d_in[8];
  const float* wv     = (const float*)d_in[9];
  const float* bv     = (const float*)d_in[10];
  const float* ln2_s  = (const float*)d_in[11];
  const float* ln2_b  = (const float*)d_in[12];
  const float* w1     = (const float*)d_in[13];
  const float* b1     = (const float*)d_in[14];
  const float* w2     = (const float*)d_in[15];
  const float* b2     = (const float*)d_in[16];
  const float* fln_s  = (const float*)d_in[17];
  const float* fln_b  = (const float*)d_in[18];
  const float* w_out  = (const float*)d_in[19];
  const float* b_out  = (const float*)d_in[20];
  float* out = (float*)d_out;

  // workspace carve-up
  char* ws = (char*)d_ws;
  size_t off = 0;
  float* xbuf = (float*)(ws + off);           off = alignup(off + (size_t)BT_ * D_ * 4);
  unsigned short* hbf = (unsigned short*)(ws + off);  off = alignup(off + (size_t)BT_ * D_ * 2);
  const long QKV_E = (long)B_ * H_ * T_ * DH_;
  unsigned short* qbf = (unsigned short*)(ws + off);  off = alignup(off + (size_t)QKV_E * 2);
  unsigned short* kbf = (unsigned short*)(ws + off);  off = alignup(off + (size_t)QKV_E * 2);
  unsigned short* vbf = (unsigned short*)(ws + off);  off = alignup(off + (size_t)QKV_E * 2);
  unsigned short* abf = (unsigned short*)(ws + off);  off = alignup(off + (size_t)BT_ * FF_ * 2);
  unsigned short* wqT = (unsigned short*)(ws + off);  off = alignup(off + (size_t)L_ * H_ * D_ * DH_ * 2);
  unsigned short* wkT = (unsigned short*)(ws + off);  off = alignup(off + (size_t)L_ * H_ * D_ * DH_ * 2);
  unsigned short* wvT = (unsigned short*)(ws + off);  off = alignup(off + (size_t)L_ * H_ * D_ * DH_ * 2);
  unsigned short* w1T = (unsigned short*)(ws + off);  off = alignup(off + (size_t)L_ * D_ * FF_ * 2);
  unsigned short* w2T = (unsigned short*)(ws + off);  off = alignup(off + (size_t)L_ * FF_ * D_ * 2);
  unsigned short* woT = (unsigned short*)(ws + off);  off = alignup(off + (size_t)D_ * V_ * 2);

  auto grid1d = [](long n) { return dim3((unsigned)((n + 255) / 256)); };

  // weight convert + transpose (f32 [.,R,C] -> bf16 [.,C,R])
  { long t = (long)L_ * H_ * D_ * DH_;
    convT_kernel<<<grid1d(t), 256, 0, stream>>>(wq, wqT, D_, DH_, t);
    convT_kernel<<<grid1d(t), 256, 0, stream>>>(wk, wkT, D_, DH_, t);
    convT_kernel<<<grid1d(t), 256, 0, stream>>>(wv, wvT, D_, DH_, t); }
  { long t = (long)L_ * D_ * FF_;
    convT_kernel<<<grid1d(t), 256, 0, stream>>>(w1, w1T, D_, FF_, t);
    convT_kernel<<<grid1d(t), 256, 0, stream>>>(w2, w2T, FF_, D_, t); }
  { long t = (long)D_ * V_;
    convT_kernel<<<grid1d(t), 256, 0, stream>>>(w_out, woT, D_, V_, t); }

  // embedding
  embed_kernel<<<grid1d((long)BT_ * D_), 256, 0, stream>>>(ctx, tok, pos, xbuf);

  const dim3 gblk(32, 4);
  for (int l = 0; l < L_; ++l) {
    // LN1 -> hbf (bf16)
    ln_kernel<<<dim3(BT_), 256, 0, stream>>>(xbuf, ln1_s + (long)l * D_, ln1_b + (long)l * D_, hbf);

    // QKV projections: per-head GEMM [BT x D] x [D x DH] -> [B,H,T,DH] bf16
    const long whl = (long)l * H_ * D_ * DH_;
    dim3 gq(1, BT_ / 16, H_);   // Ntiles = DH/16 = 4 (one block of 4 waves)
    gemm_bf16_wmma<<<gq, gblk, 0, stream>>>(hbf, D_, wqT + whl, D_, (long)D_ * DH_,
        bq + (long)l * H_ * DH_, DH_, nullptr, qbf,
        (long)H_ * T_ * DH_, DH_, (long)T_ * DH_, BT_, DH_, D_, T_, 0, 0);
    gemm_bf16_wmma<<<gq, gblk, 0, stream>>>(hbf, D_, wkT + whl, D_, (long)D_ * DH_,
        bk + (long)l * H_ * DH_, DH_, nullptr, kbf,
        (long)H_ * T_ * DH_, DH_, (long)T_ * DH_, BT_, DH_, D_, T_, 0, 0);
    gemm_bf16_wmma<<<gq, gblk, 0, stream>>>(hbf, D_, wvT + whl, D_, (long)D_ * DH_,
        bv + (long)l * H_ * DH_, DH_, nullptr, vbf,
        (long)H_ * T_ * DH_, DH_, (long)T_ * DH_, BT_, DH_, D_, T_, 0, 0);

    // attention: x += softmax(QK^T)V  (heads concatenated)
    attn_kernel<<<dim3(T_ / 16, B_ * H_), dim3(32), 0, stream>>>(qbf, kbf, vbf, xbuf);

    // LN2 -> hbf
    ln_kernel<<<dim3(BT_), 256, 0, stream>>>(xbuf, ln2_s + (long)l * D_, ln2_b + (long)l * D_, hbf);

    // FFN up: a = relu(h @ w1 + b1) -> bf16 [BT, FF]
    gemm_bf16_wmma<<<dim3(FF_ / 16 / 4, BT_ / 16, 1), gblk, 0, stream>>>(
        hbf, D_, w1T + (long)l * D_ * FF_, D_, 0,
        b1 + (long)l * FF_, 0, nullptr, abf,
        0, FF_, 0, BT_, FF_, D_, BT_, 1, 0);

    // FFN down + residual: x += a @ w2 + b2
    gemm_bf16_wmma<<<dim3(D_ / 16 / 4, BT_ / 16, 1), gblk, 0, stream>>>(
        abf, FF_, w2T + (long)l * FF_ * D_, FF_, 0,
        b2 + (long)l * D_, 0, xbuf, nullptr,
        0, D_, 0, BT_, D_, FF_, BT_, 0, 1);
  }

  // final LN -> hbf, then logits: out = h @ w_out + b_out  (f32 [B,T,V])
  ln_kernel<<<dim3(BT_), 256, 0, stream>>>(xbuf, fln_s, fln_b, hbf);
  gemm_bf16_wmma<<<dim3(V_ / 16 / 4, BT_ / 16, 1), gblk, 0, stream>>>(
      hbf, D_, woT, D_, 0,
      b_out, 0, out, nullptr,
      0, V_, 0, BT_, V_, D_, BT_, 0, 0);
}
